// GPLinear_72447508349460
// MI455X (gfx1250) — matrix operand order repaired and checked by reading, not compiled, not run on hardware
//
#include <hip/hip_runtime.h>

// ---------------------------------------------------------------------------
// GPLinear (Cl(4,0) geometric-product linear layer) for MI455X / gfx1250.
//
//   out[b,o,k] = sum_{p,j} x[b,p,j] * Wg[p,j,o,k] + bias[o,k]
//   Wg[p,j,o,k] = GP[j^k, j, k] * W[p,o,j^k]      (GP entries are +-1)
//
// => single GEMM  M=16384 x K=1024 x N=1024 (34.4 GFLOP, ~132 MB traffic).
// bf16 WMMA (v_wmma_f32_16x16x32_bf16) with fp32 accumulate keeps compute at
// the 23.3 TB/s HBM floor (~6 us). B tiles (pre-folded bf16 weights) are
// staged with GLOBAL_LOAD_ASYNC_TO_LDS_B128 (ASYNCcnt path); A tiles are
// converted fp32->bf16 in-flight with v_cvt_pk_bf16_f32.
// ---------------------------------------------------------------------------

typedef __attribute__((ext_vector_type(16))) __bf16       v16bf;
typedef __attribute__((ext_vector_type(8)))  float        v8f;
typedef __attribute__((ext_vector_type(4)))  float        v4f;
typedef __attribute__((ext_vector_type(4)))  unsigned int v4u;

#define BATCH 16384
#define NIN   64
#define NOUT  64
#define BD    16
#define K_DIM (NIN * BD)   // 1024
#define N_DIM (NOUT * BD)  // 1024

__device__ __forceinline__ unsigned short f32_to_bf16(float f) {
  unsigned u = __float_as_uint(f);
  unsigned r = u + 0x7FFFu + ((u >> 16) & 1u);   // round-to-nearest-even
  return (unsigned short)(r >> 16);
}

// v_cvt_pk_bf16_f32: dst[15:0]=bf16(lo), dst[31:16]=bf16(hi), RNE (default mode)
__device__ __forceinline__ unsigned pack_bf16(float lo, float hi) {
  unsigned r;
  asm("v_cvt_pk_bf16_f32 %0, %1, %2" : "=v"(r) : "v"(lo), "v"(hi));
  return r;
}

// Flat address of a __shared__ object is {SHARED_BASE[63:32], lds_offset[31:0]}
// (ISA 10.2), so the low 32 bits are the DS byte address.
__device__ __forceinline__ unsigned lds_u32(const void* p) {
  return (unsigned)(unsigned long long)p;
}

// ---------------------------------------------------------------------------
// Kernel 1: fold Cayley table into W, store bf16 transposed: WgT[n=o*16+k][r=p*16+j]
// ---------------------------------------------------------------------------
__global__ __launch_bounds__(256) void gp_fold_w(const float* __restrict__ W,
                                                 const float* __restrict__ GP,
                                                 unsigned short* __restrict__ WgT) {
  int tid = blockIdx.x * 256 + threadIdx.x;     // 0 .. 1M-1,  tid = c*1024 + r
  int r = tid & (K_DIM - 1);
  int c = tid >> 10;
  int p = r >> 4, j = r & 15;
  int o = c >> 4, kb = c & 15;
  int i = j ^ kb;
  float s = GP[(i * 16 + j) * 16 + kb];         // +-1
  float w = W[(p * NOUT + o) * BD + i];
  WgT[tid] = f32_to_bf16(s * w);
}

// ---------------------------------------------------------------------------
// Kernel 2: GEMM  Out[M,N] = X[M,K] * WgT^T + bias,  bf16 WMMA, fp32 accum.
// ---------------------------------------------------------------------------
#define BM 128
#define BN 128
#define BK 32
#define LDK 40          // LDS row stride (bf16 elems): 80 B, 16B-aligned, conflict-free
#define NKSTEP (K_DIM / BK)  // 32

union AFrag { v16bf v; v4u q[2]; };

__device__ __forceinline__ void stage_a(unsigned short* dstA,
                                        v4f a0, v4f a1, v4f a2, v4f a3) {
  v4u pa0 = { pack_bf16(a0[0], a0[1]), pack_bf16(a0[2], a0[3]),
              pack_bf16(a1[0], a1[1]), pack_bf16(a1[2], a1[3]) };
  v4u pa1 = { pack_bf16(a2[0], a2[1]), pack_bf16(a2[2], a2[3]),
              pack_bf16(a3[0], a3[1]), pack_bf16(a3[2], a3[3]) };
  *(v4u*)(dstA)     = pa0;
  *(v4u*)(dstA + 8) = pa1;
}

// Async copy 32 bytes global -> LDS (two b128 ops; offset applies to both sides).
__device__ __forceinline__ void async_copy_32B(unsigned ldsAddr, const void* gaddr) {
  asm volatile(
      "global_load_async_to_lds_b128 %0, %1, off\n\t"
      "global_load_async_to_lds_b128 %0, %1, off offset:16"
      :: "v"(ldsAddr), "v"((unsigned long long)gaddr) : "memory");
}
__device__ __forceinline__ void wait_async0() {
  asm volatile("s_wait_asynccnt 0x0" ::: "memory");
}

__global__ __launch_bounds__(256)
void gp_linear_gemm(const float* __restrict__ X,
                    const unsigned short* __restrict__ WgT,
                    const float* __restrict__ Bias,
                    float* __restrict__ Out) {
  __shared__ __align__(16) unsigned short As[2][BM * LDK];  // bf16 tiles, 20 KB
  __shared__ __align__(16) unsigned short Bs[2][BN * LDK];  // bf16 tiles, 20 KB

  const int tid   = threadIdx.x;
  const int lane  = tid & 31;
  const int wave  = tid >> 5;     // 8 waves
  const int waveM = wave >> 2;    // 0..1 : 64 rows each
  const int waveN = wave & 3;     // 0..3 : 32 cols each
  const int hf    = lane >> 4;    // lane half (wave32)
  const int lr    = lane & 15;

  const int mBase = blockIdx.x * BM;
  const int nBase = blockIdx.y * BN;

  // staging coords: 2 threads per row, 16 elems (64B of fp32 / 32B of bf16) each
  const int sRow = tid >> 1;            // 0..127
  const int sCol = (tid & 1) * 16;      // 0 or 16

  const float*          gA = X   + (size_t)(mBase + sRow) * K_DIM + sCol;
  const unsigned short* gB = WgT + (size_t)(nBase + sRow) * K_DIM + sCol;

  v8f acc[4][2] = {};

  float biasv[2];
  biasv[0] = Bias[nBase + waveN * 32 + 0 * 16 + lr];
  biasv[1] = Bias[nBase + waveN * 32 + 1 * 16 + lr];

  // ---- prologue: stage k-tile 0 into buffer 0
  {
    async_copy_32B(lds_u32(&Bs[0][sRow * LDK + sCol]), gB);
    v4f a0 = *(const v4f*)(gA + 0);
    v4f a1 = *(const v4f*)(gA + 4);
    v4f a2 = *(const v4f*)(gA + 8);
    v4f a3 = *(const v4f*)(gA + 12);
    stage_a(&As[0][sRow * LDK + sCol], a0, a1, a2, a3);
    wait_async0();
  }
  __syncthreads();

  for (int kt = 0; kt < NKSTEP; ++kt) {
    const int cur = kt & 1, nxt = cur ^ 1;

    // ---- prefetch next k-tile: B straight to LDS (async), A into registers
    v4f a0, a1, a2, a3;
    if (kt + 1 < NKSTEP) {
      async_copy_32B(lds_u32(&Bs[nxt][sRow * LDK + sCol]), gB + (kt + 1) * BK);
      const float* pA = gA + (kt + 1) * BK;
      a0 = *(const v4f*)(pA + 0);
      a1 = *(const v4f*)(pA + 4);
      a2 = *(const v4f*)(pA + 8);
      a3 = *(const v4f*)(pA + 12);
    }

    // ---- load fragments from LDS (2x ds_load_b128 per fragment)
    AFrag af[4], bfr[2];
#pragma unroll
    for (int mf = 0; mf < 4; ++mf) {
      const unsigned short* ar = &As[cur][(waveM * 64 + mf * 16 + lr) * LDK];
      // A 16x32 bf16 layout: VGPR v<4 -> K = hf*8 + 2v ; v>=4 -> K = 16 + hf*8 + 2(v-4)
      af[mf].q[0] = *(const v4u*)((const char*)ar + hf * 16);
      af[mf].q[1] = *(const v4u*)((const char*)ar + 32 + hf * 16);
    }
#pragma unroll
    for (int nf = 0; nf < 2; ++nf) {
      const unsigned short* br = &Bs[cur][(waveN * 32 + nf * 16 + lr) * LDK];
      // B 32x16 bf16 layout: VGPR v<4 -> K = hf*8 + 2v ; v>=4 -> K = 16 + hf*8 + 2(v-4)
      bfr[nf].q[0] = *(const v4u*)((const char*)br + hf * 16);
      bfr[nf].q[1] = *(const v4u*)((const char*)br + 32 + hf * 16);
    }

    // ---- 8 WMMAs per wave per k-step (64 per block: 128x128x32)
#pragma unroll
    for (int mf = 0; mf < 4; ++mf)
#pragma unroll
      for (int nf = 0; nf < 2; ++nf)
        acc[mf][nf] = __builtin_amdgcn_wmma_f32_16x16x32_bf16(
            false, af[mf].v, false, bfr[nf].v, (short)0, acc[mf][nf],
            false, false);

    // ---- commit prefetched A tile; drain our async B copies, then barrier
    if (kt + 1 < NKSTEP) {
      stage_a(&As[nxt][sRow * LDK + sCol], a0, a1, a2, a3);
      wait_async0();
    }
    __syncthreads();
  }

  // ---- epilogue: C layout (16x16 f32): elem(m = v + 8*hf, n = lr) in VGPR v
#pragma unroll
  for (int mf = 0; mf < 4; ++mf) {
#pragma unroll
    for (int nf = 0; nf < 2; ++nf) {
      const int n = nBase + waveN * 32 + nf * 16 + lr;
#pragma unroll
      for (int v = 0; v < 8; ++v) {
        const int m = mBase + waveM * 64 + mf * 16 + hf * 8 + v;
        Out[(size_t)m * N_DIM + n] = acc[mf][nf][v] + biasv[nf];
      }
    }
  }
}

// ---------------------------------------------------------------------------
extern "C" void kernel_launch(void* const* d_in, const int* in_sizes, int n_in,
                              void* d_out, int out_size, void* d_ws, size_t ws_size,
                              hipStream_t stream) {
  (void)in_sizes; (void)n_in; (void)out_size; (void)ws_size;
  const float* x  = (const float*)d_in[0];   // (16384, 64, 16) f32
  const float* W  = (const float*)d_in[1];   // (64, 64, 16)    f32
  const float* b  = (const float*)d_in[2];   // (64, 16)        f32
  const float* GP = (const float*)d_in[3];   // (16, 16, 16)    f32 (+-1 sparse)

  unsigned short* WgT = (unsigned short*)d_ws;  // needs 2 MiB scratch (bf16 1024x1024)

  gp_fold_w<<<(K_DIM * N_DIM) / 256, 256, 0, stream>>>(W, GP, WgT);

  dim3 grid(BATCH / BM, N_DIM / BN);  // 128 x 8 workgroups
  gp_linear_gemm<<<grid, 256, 0, stream>>>(x, WgT, b, (float*)d_out);
}
